// GMN_63574105915539
// MI455X (gfx1250) — matrix operand order repaired and checked by reading, not compiled, not run on hardware
//
#include <hip/hip_runtime.h>

#define N_NODES 50000
#define N_EDGES 800000
#define L 10

typedef __bf16 bf16;
typedef __attribute__((ext_vector_type(16))) __bf16 v16bf;
typedef __attribute__((ext_vector_type(8)))  __bf16 v8bf;
typedef __attribute__((ext_vector_type(4)))  __bf16 v4bf;
typedef __attribute__((ext_vector_type(8)))  float  v8f;
typedef __attribute__((ext_vector_type(4)))  float  v4f;

#define ROWS 128      // node rows per block
#define THREADS 128   // 4 waves (wave32), 32 rows per wave (2 x 16-row slabs)

// ---------------------------------------------------------------------------
// Weight swizzle: fp32 [K,128] row-major -> bf16, WMMA B-fragment order.
//   lane = (half<<4)|lanelo ; element i of v16bf -> K = (i&8)*2 + half*8 + (i&7), N = lanelo
//   dst index = (((ct*KB + kb)*32 + lane)*16 + i)
// ---------------------------------------------------------------------------
__global__ void swz_kernel(const float* __restrict__ src, bf16* __restrict__ dst,
                           int K) {
  int KB = K >> 5;
  int total = 4096 * KB;
  int t = blockIdx.x * blockDim.x + threadIdx.x;
  if (t >= total) return;
  int i      = t & 15;
  int lane   = (t >> 4) & 31;
  int r      = t >> 9;
  int kb     = r % KB;
  int ct     = r / KB;
  int lanelo = lane & 15;
  int half   = lane >> 4;
  int k   = kb * 32 + (i & 8) * 2 + half * 8 + (i & 7);
  int col = ct * 16 + lanelo;
  dst[t] = (bf16)src[k * 128 + col];
}

// ---------------------------------------------------------------------------
// One wave computes a 32-row x 128-col GEMM slab pair; each B fragment is
// reused for two WMMAs (halves B traffic vs one-slab version).
// A: LDS row-major bf16 [32 x K]; slab0 rows 0..15, slab1 rows 16..31.
// ---------------------------------------------------------------------------
__device__ static inline void wave_gemm2(const bf16* __restrict__ A, int lda, int KB,
                                         const bf16* __restrict__ Bsw,
                                         v8f acc0[8], v8f acc1[8]) {
  int lane   = threadIdx.x & 31;
  int lanelo = lane & 15;
  int half   = lane >> 4;
  const bf16* arow0 = A + lanelo * lda;
  const bf16* arow1 = A + (16 + lanelo) * lda;
  for (int kb = 0; kb < KB; ++kb) {
    v8bf a0lo = *(const v8bf*)(arow0 + kb * 32 + half * 8);
    v8bf a0hi = *(const v8bf*)(arow0 + kb * 32 + 16 + half * 8);
    v8bf a1lo = *(const v8bf*)(arow1 + kb * 32 + half * 8);
    v8bf a1hi = *(const v8bf*)(arow1 + kb * 32 + 16 + half * 8);
    v16bf a0, a1;
#pragma unroll
    for (int q = 0; q < 8; ++q) {
      a0[q] = a0lo[q]; a0[q + 8] = a0hi[q];
      a1[q] = a1lo[q]; a1[q + 8] = a1hi[q];
    }
#pragma unroll
    for (int ct = 0; ct < 8; ++ct) {
      v16bf b = *(const v16bf*)(Bsw + (((ct * KB + kb) * 32 + lane) << 4));
      acc0[ct] = __builtin_amdgcn_wmma_f32_16x16x32_bf16(
          false, a0, false, b, (short)0, acc0[ct], false, false);
      acc1[ct] = __builtin_amdgcn_wmma_f32_16x16x32_bf16(
          false, a1, false, b, (short)0, acc1[ct], false, false);
    }
  }
}

__device__ static inline void acc_zero(v8f acc[8]) {
#pragma unroll
  for (int c = 0; c < 8; ++c) {
    v8f z = {0.f, 0.f, 0.f, 0.f, 0.f, 0.f, 0.f, 0.f};
    acc[c] = z;
  }
}

// C-tile layout: VGPR j -> row (half*8 + j), col = ct*16 + lanelo
__device__ static inline void epi_relu_bf16(const v8f acc[8],
                                            const float* __restrict__ bias,
                                            bf16* __restrict__ out, int ldo) {
  int lane = threadIdx.x & 31, lanelo = lane & 15, half = lane >> 4;
#pragma unroll
  for (int ct = 0; ct < 8; ++ct) {
    float b = bias[ct * 16 + lanelo];
#pragma unroll
    for (int j = 0; j < 8; ++j) {
      float v = acc[ct][j] + b;
      out[(half * 8 + j) * ldo + ct * 16 + lanelo] = (bf16)(v > 0.f ? v : 0.f);
    }
  }
}

// Fused bias + LayerNorm + store-to-global epilogue for the final GEMM of an
// MLP block. Row r's 128 cols live in 16 lanes of one half-wave, 8 regs each;
// mean/var reduce with 4 shfl_xor steps (masks 1/2/4/8 stay inside the half).
__device__ static inline void epi_ln_store(v8f acc[8],
                                           const float* __restrict__ bias,
                                           const float* __restrict__ g,
                                           const float* __restrict__ be,
                                           float* __restrict__ h, int gbase) {
  int lane = threadIdx.x & 31, lanelo = lane & 15, half = lane >> 4;
  float gv[8], bv[8];
#pragma unroll
  for (int ct = 0; ct < 8; ++ct) {
    int col = ct * 16 + lanelo;
    float b = bias[col];
    gv[ct] = g[col];
    bv[ct] = be[col];
#pragma unroll
    for (int j = 0; j < 8; ++j) acc[ct][j] += b;
  }
#pragma unroll
  for (int j = 0; j < 8; ++j) {
    float s = 0.f;
#pragma unroll
    for (int ct = 0; ct < 8; ++ct) s += acc[ct][j];
    for (int m = 1; m < 16; m <<= 1) s += __shfl_xor(s, m, 32);
    float mu = s * (1.f / 128.f);
    float v = 0.f;
#pragma unroll
    for (int ct = 0; ct < 8; ++ct) { float d = acc[ct][j] - mu; v += d * d; }
    for (int m = 1; m < 16; m <<= 1) v += __shfl_xor(v, m, 32);
    float inv = rsqrtf(v * (1.f / 128.f) + 1e-5f);
    int gr = gbase + half * 8 + j;
    if (gr < N_NODES) {
      float* dst = h + (size_t)gr * 128;
#pragma unroll
      for (int ct = 0; ct < 8; ++ct)
        dst[ct * 16 + lanelo] = (acc[ct][j] - mu) * inv * gv[ct] + bv[ct];
    }
  }
}

// ---------------------------------------------------------------------------
// Encoder: h = LN(MLP4(x));  layer1 (12->128) scalar, layers 2..4 WMMA.
// ---------------------------------------------------------------------------
__global__ void __launch_bounds__(THREADS, 1)
enc_kernel(const float* __restrict__ x, float* __restrict__ h,
           const float* __restrict__ W1, const float* __restrict__ b1,
           const bf16* __restrict__ W2s, const float* __restrict__ b2,
           const bf16* __restrict__ W3s, const float* __restrict__ b3,
           const bf16* __restrict__ W4s, const float* __restrict__ b4,
           const float* __restrict__ g, const float* __restrict__ be) {
  __shared__ __align__(16) bf16  sB[ROWS * 128];
  __shared__ __align__(16) bf16  sC[ROWS * 128];
  __shared__ float sX[ROWS * 12];
  __shared__ float sW1[12 * 128];
  int tid = threadIdx.x;
  int r0 = blockIdx.x * ROWS;

  for (int idx = tid; idx < ROWS * 12; idx += THREADS) {
    int row = idx / 12, col = idx % 12;
    int gr = r0 + row;
    sX[idx] = (gr < N_NODES) ? x[(size_t)gr * 12 + col] : 0.f;
  }
  for (int idx = tid; idx < 12 * 128; idx += THREADS) sW1[idx] = W1[idx];
  __syncthreads();

  // layer1: relu(x @ W1 + b1) -> sB (bf16)
  for (int idx = tid; idx < ROWS * 128; idx += THREADS) {
    int row = idx >> 7, col = idx & 127;
    float s = b1[col];
#pragma unroll
    for (int k = 0; k < 12; ++k) s += sX[row * 12 + k] * sW1[k * 128 + col];
    sB[idx] = (bf16)(s > 0.f ? s : 0.f);
  }
  __syncthreads();

  int wave = tid >> 5;
  int wr = wave * 32;
  v8f acc0[8], acc1[8];
  acc_zero(acc0); acc_zero(acc1);
  wave_gemm2(sB + wr * 128, 128, 4, W2s, acc0, acc1);
  epi_relu_bf16(acc0, b2, sC + wr * 128, 128);
  epi_relu_bf16(acc1, b2, sC + (wr + 16) * 128, 128);
  __syncthreads();
  acc_zero(acc0); acc_zero(acc1);
  wave_gemm2(sC + wr * 128, 128, 4, W3s, acc0, acc1);
  epi_relu_bf16(acc0, b3, sB + wr * 128, 128);
  epi_relu_bf16(acc1, b3, sB + (wr + 16) * 128, 128);
  __syncthreads();
  acc_zero(acc0); acc_zero(acc1);
  wave_gemm2(sB + wr * 128, 128, 4, W4s, acc0, acc1);
  epi_ln_store(acc0, b4, g, be, h, r0 + wr);
  epi_ln_store(acc1, b4, g, be, h, r0 + wr + 16);
}

// ---------------------------------------------------------------------------
// Message-passing layer: h = LN(MLP4([h, agg]))
// ---------------------------------------------------------------------------
__global__ void __launch_bounds__(THREADS, 1)
mp_kernel(float* __restrict__ h, const float* __restrict__ agg,
          const bf16* __restrict__ W1s, const float* __restrict__ b1,
          const bf16* __restrict__ W2s, const float* __restrict__ b2,
          const bf16* __restrict__ W3s, const float* __restrict__ b3,
          const bf16* __restrict__ W4s, const float* __restrict__ b4,
          const float* __restrict__ g, const float* __restrict__ be) {
  __shared__ __align__(16) bf16 sA[ROWS * 256];   // cat input, reused 128x128 later
  __shared__ __align__(16) bf16 sB[ROWS * 128];
  int tid = threadIdx.x;
  int r0 = blockIdx.x * ROWS;

  // stage cat = [h, agg] as bf16, 4 cols per thread (float4 -> 4x bf16)
  for (int q = tid; q < ROWS * 64; q += THREADS) {
    int row = q >> 6;
    int c4  = (q & 63) * 4;          // 0..252, never straddles the 128 split
    int gr = r0 + row;
    v4f v = {0.f, 0.f, 0.f, 0.f};
    if (gr < N_NODES) {
      const float* src = (c4 < 128) ? (h + (size_t)gr * 128 + c4)
                                    : (agg + (size_t)gr * 128 + (c4 - 128));
      v = *(const v4f*)src;
    }
    v4bf o; o[0] = (bf16)v[0]; o[1] = (bf16)v[1]; o[2] = (bf16)v[2]; o[3] = (bf16)v[3];
    *(v4bf*)(sA + row * 256 + c4) = o;
  }
  __syncthreads();

  int wave = tid >> 5;
  int wr = wave * 32;
  v8f acc0[8], acc1[8];
  acc_zero(acc0); acc_zero(acc1);
  wave_gemm2(sA + wr * 256, 256, 8, W1s, acc0, acc1);       // 256 -> 128
  epi_relu_bf16(acc0, b1, sB + wr * 128, 128);
  epi_relu_bf16(acc1, b1, sB + (wr + 16) * 128, 128);
  __syncthreads();
  acc_zero(acc0); acc_zero(acc1);
  wave_gemm2(sB + wr * 128, 128, 4, W2s, acc0, acc1);
  epi_relu_bf16(acc0, b2, sA + wr * 128, 128);              // reuse sA as 128x128
  epi_relu_bf16(acc1, b2, sA + (wr + 16) * 128, 128);
  __syncthreads();
  acc_zero(acc0); acc_zero(acc1);
  wave_gemm2(sA + wr * 128, 128, 4, W3s, acc0, acc1);
  epi_relu_bf16(acc0, b3, sB + wr * 128, 128);
  epi_relu_bf16(acc1, b3, sB + (wr + 16) * 128, 128);
  __syncthreads();
  acc_zero(acc0); acc_zero(acc1);
  wave_gemm2(sB + wr * 128, 128, 4, W4s, acc0, acc1);
  epi_ln_store(acc0, b4, g, be, h, r0 + wr);
  epi_ln_store(acc1, b4, g, be, h, r0 + wr + 16);
}

// ---------------------------------------------------------------------------
// Decoder: out[:,1:3] of MLP4(h) (no LN); layer4 (128->12) scalar, cols 1..2.
// ---------------------------------------------------------------------------
__global__ void __launch_bounds__(THREADS, 1)
dec_kernel(const float* __restrict__ h, float* __restrict__ out,
           const bf16* __restrict__ W1s, const float* __restrict__ b1,
           const bf16* __restrict__ W2s, const float* __restrict__ b2,
           const bf16* __restrict__ W3s, const float* __restrict__ b3,
           const float* __restrict__ W4, const float* __restrict__ b4) {
  __shared__ __align__(16) bf16 sA[ROWS * 128];
  __shared__ __align__(16) bf16 sB[ROWS * 128];
  int tid = threadIdx.x;
  int r0 = blockIdx.x * ROWS;

  for (int q = tid; q < ROWS * 32; q += THREADS) {
    int row = q >> 5;
    int c4  = (q & 31) * 4;
    int gr = r0 + row;
    v4f v = {0.f, 0.f, 0.f, 0.f};
    if (gr < N_NODES) v = *(const v4f*)(h + (size_t)gr * 128 + c4);
    v4bf o; o[0] = (bf16)v[0]; o[1] = (bf16)v[1]; o[2] = (bf16)v[2]; o[3] = (bf16)v[3];
    *(v4bf*)(sA + row * 128 + c4) = o;
  }
  __syncthreads();

  int wave = tid >> 5;
  int wr = wave * 32;
  v8f acc0[8], acc1[8];
  acc_zero(acc0); acc_zero(acc1);
  wave_gemm2(sA + wr * 128, 128, 4, W1s, acc0, acc1);
  epi_relu_bf16(acc0, b1, sB + wr * 128, 128);
  epi_relu_bf16(acc1, b1, sB + (wr + 16) * 128, 128);
  __syncthreads();
  acc_zero(acc0); acc_zero(acc1);
  wave_gemm2(sB + wr * 128, 128, 4, W2s, acc0, acc1);
  epi_relu_bf16(acc0, b2, sA + wr * 128, 128);
  epi_relu_bf16(acc1, b2, sA + (wr + 16) * 128, 128);
  __syncthreads();
  acc_zero(acc0); acc_zero(acc1);
  wave_gemm2(sA + wr * 128, 128, 4, W3s, acc0, acc1);
  epi_relu_bf16(acc0, b3, sB + wr * 128, 128);
  epi_relu_bf16(acc1, b3, sB + (wr + 16) * 128, 128);
  __syncthreads();

  // layer4: each thread does one row, output columns 1 and 2 only
  if (tid < ROWS) {
    int gr = r0 + tid;
    if (gr < N_NODES) {
      float s1 = b4[1], s2 = b4[2];
      const bf16* arow = sB + tid * 128;
      for (int k = 0; k < 128; ++k) {
        float a = (float)arow[k];
        s1 += a * W4[k * 12 + 1];
        s2 += a * W4[k * 12 + 2];
      }
      out[(size_t)gr * 2 + 0] = s1;
      out[(size_t)gr * 2 + 1] = s2;
    }
  }
}

// ---------------------------------------------------------------------------
// Edge scatter: agg[recv] += h[send]; one wave per edge, 4 f32 atomics/lane.
// ---------------------------------------------------------------------------
__global__ void zero_kernel(float* __restrict__ p, int n) {
  int i = blockIdx.x * blockDim.x + threadIdx.x;
  if (i < n) p[i] = 0.f;
}

__global__ void scatter_kernel(const float* __restrict__ h,
                               const int* __restrict__ senders,
                               const int* __restrict__ receivers,
                               float* __restrict__ agg) {
  int t = blockIdx.x * blockDim.x + threadIdx.x;
  int e = t >> 5;
  int lane = t & 31;
  if (e >= N_EDGES) return;
  int s = senders[e], r = receivers[e];
  const float* src = h + (size_t)s * 128;
  float* dst = agg + (size_t)r * 128;
#pragma unroll
  for (int j = 0; j < 4; ++j)
    atomicAdd(dst + lane + 32 * j, src[lane + 32 * j]);
}

// ---------------------------------------------------------------------------
extern "C" void kernel_launch(void* const* d_in, const int* in_sizes, int n_in,
                              void* d_out, int out_size, void* d_ws, size_t ws_size,
                              hipStream_t stream) {
  (void)in_sizes; (void)n_in; (void)out_size; (void)ws_size;
  const float* x         = (const float*)d_in[0];
  const int*   senders   = (const int*)d_in[1];
  const int*   receivers = (const int*)d_in[2];
  const float* enc_W1 = (const float*)d_in[3];  const float* enc_b1 = (const float*)d_in[4];
  const float* enc_W2 = (const float*)d_in[5];  const float* enc_b2 = (const float*)d_in[6];
  const float* enc_W3 = (const float*)d_in[7];  const float* enc_b3 = (const float*)d_in[8];
  const float* enc_W4 = (const float*)d_in[9];  const float* enc_b4 = (const float*)d_in[10];
  const float* enc_g  = (const float*)d_in[11]; const float* enc_be = (const float*)d_in[12];
  const float* mp_W1  = (const float*)d_in[13]; const float* mp_b1  = (const float*)d_in[14];
  const float* mp_W2  = (const float*)d_in[15]; const float* mp_b2  = (const float*)d_in[16];
  const float* mp_W3  = (const float*)d_in[17]; const float* mp_b3  = (const float*)d_in[18];
  const float* mp_W4  = (const float*)d_in[19]; const float* mp_b4  = (const float*)d_in[20];
  const float* mp_g   = (const float*)d_in[21]; const float* mp_be  = (const float*)d_in[22];
  const float* dec_W1 = (const float*)d_in[23]; const float* dec_b1 = (const float*)d_in[24];
  const float* dec_W2 = (const float*)d_in[25]; const float* dec_b2 = (const float*)d_in[26];
  const float* dec_W3 = (const float*)d_in[27]; const float* dec_b3 = (const float*)d_in[28];
  const float* dec_W4 = (const float*)d_in[29]; const float* dec_b4 = (const float*)d_in[30];

  // ---- workspace carve ----
  char* ws = (char*)d_ws;
  size_t off = 0;
  auto carve = [&](size_t bytes) -> void* {
    void* p = ws + off;
    off = (off + bytes + 255) & ~(size_t)255;
    return p;
  };
  float* h   = (float*)carve((size_t)N_NODES * 128 * sizeof(float));
  float* agg = (float*)carve((size_t)N_NODES * 128 * sizeof(float));
  bf16* encW2s = (bf16*)carve(16384 * sizeof(bf16));
  bf16* encW3s = (bf16*)carve(16384 * sizeof(bf16));
  bf16* encW4s = (bf16*)carve(16384 * sizeof(bf16));
  bf16* mpW1s  = (bf16*)carve((size_t)L * 32768 * sizeof(bf16));
  bf16* mpW2s  = (bf16*)carve((size_t)L * 16384 * sizeof(bf16));
  bf16* mpW3s  = (bf16*)carve((size_t)L * 16384 * sizeof(bf16));
  bf16* mpW4s  = (bf16*)carve((size_t)L * 16384 * sizeof(bf16));
  bf16* decW1s = (bf16*)carve(16384 * sizeof(bf16));
  bf16* decW2s = (bf16*)carve(16384 * sizeof(bf16));
  bf16* decW3s = (bf16*)carve(16384 * sizeof(bf16));

  auto swz = [&](const float* src, bf16* dst, int K) {
    int total = 4096 * (K >> 5);
    swz_kernel<<<(total + 255) / 256, 256, 0, stream>>>(src, dst, K);
  };
  swz(enc_W2, encW2s, 128);
  swz(enc_W3, encW3s, 128);
  swz(enc_W4, encW4s, 128);
  for (int l = 0; l < L; ++l) {
    swz(mp_W1 + (size_t)l * 32768, mpW1s + (size_t)l * 32768, 256);
    swz(mp_W2 + (size_t)l * 16384, mpW2s + (size_t)l * 16384, 128);
    swz(mp_W3 + (size_t)l * 16384, mpW3s + (size_t)l * 16384, 128);
    swz(mp_W4 + (size_t)l * 16384, mpW4s + (size_t)l * 16384, 128);
  }
  swz(dec_W1, decW1s, 128);
  swz(dec_W2, decW2s, 128);
  swz(dec_W3, decW3s, 128);

  int nblk = (N_NODES + ROWS - 1) / ROWS;

  enc_kernel<<<nblk, THREADS, 0, stream>>>(x, h, enc_W1, enc_b1,
      encW2s, enc_b2, encW3s, enc_b3, encW4s, enc_b4, enc_g, enc_be);

  for (int l = 0; l < L; ++l) {
    int nE = N_NODES * 128;
    zero_kernel<<<(nE + 255) / 256, 256, 0, stream>>>(agg, nE);
    long long tE = (long long)N_EDGES * 32;
    scatter_kernel<<<(int)((tE + 255) / 256), 256, 0, stream>>>(h, senders, receivers, agg);
    mp_kernel<<<nblk, THREADS, 0, stream>>>(h, agg,
        mpW1s + (size_t)l * 32768, mp_b1 + (size_t)l * 128,
        mpW2s + (size_t)l * 16384, mp_b2 + (size_t)l * 128,
        mpW3s + (size_t)l * 16384, mp_b3 + (size_t)l * 128,
        mpW4s + (size_t)l * 16384, mp_b4 + (size_t)l * 128,
        mp_g + (size_t)l * 128, mp_be + (size_t)l * 128);
  }

  dec_kernel<<<nblk, THREADS, 0, stream>>>(h, (float*)d_out,
      decW1s, dec_b1, decW2s, dec_b2, decW3s, dec_b3, dec_W4, dec_b4);
}
